// MLP_Adapter_slide8_1726576853561
// MI455X (gfx1250) — matrix operand order repaired and checked by reading, not compiled, not run on hardware
//
#include <hip/hip_runtime.h>
#include <math.h>

// ---------------------------------------------------------------------------
// CDNA5 (gfx1250) WMMA helpers: bf16 16x16x32, wave32.
// All WMMA operands live in LDS as packed bf16 so fragment builds are pure
// ds_load_b32 of pre-packed pairs (no VALU conversion on the critical path).
// ---------------------------------------------------------------------------
typedef __attribute__((ext_vector_type(16))) __bf16        v16bf;
typedef __attribute__((ext_vector_type(8)))  float         v8f;
typedef __attribute__((ext_vector_type(8)))  unsigned int  v8u;

// cheap round-to-nearest fp32 -> bf16 (single element)
__device__ __forceinline__ unsigned short f2bf16(float f) {
  return (unsigned short)((__float_as_uint(f) + 0x8000u) >> 16);
}
// pack two fp32 -> bf16x2 in one u32 (4 VALU)
__device__ __forceinline__ unsigned int packbf2(float lo, float hi) {
  unsigned int a = (__float_as_uint(lo) + 0x8000u) >> 16;
  unsigned int b = (__float_as_uint(hi) + 0x8000u) & 0xffff0000u;
  return a | b;
}

// A fragment (16x32 bf16) from bf16 LDS rows [16][stride_h] (stride_h even).
// ISA 7.12.2: lane L row M=L&15; lanes>=16 add K+8;
// VGPR0..3 -> K pairs {0,2,4,6}, VGPR4..7 -> K pairs {16,18,20,22}.
__device__ __forceinline__ v16bf load_a_bf16(const unsigned short* rows, int stride,
                                             int lane, int kc) {
  const unsigned short* rp = rows + (lane & 15) * stride + kc + ((lane >> 4) << 3);
  v8u u;
#pragma unroll
  for (int r = 0; r < 8; ++r)
    u[r] = *(const unsigned int*)(rp + ((r & 3) << 1) + ((r >> 2) << 4));
  return __builtin_bit_cast(v16bf, u);
}

// B fragment (32x16 bf16) from bf16 LDS stored row-major [N][K] (stride even).
// lane L col N=L&15; lanes>=16 add K+8; VGPR0..3 -> K pairs {0,2,4,6}, VGPR4..7 -> {16,18,20,22}.
__device__ __forceinline__ v16bf load_b_bf16(const unsigned short* w, int stride, int nbase,
                                             int lane, int kc) {
  const unsigned short* rp = w + (nbase + (lane & 15)) * stride + kc + ((lane >> 4) << 3);
  v8u u;
#pragma unroll
  for (int r = 0; r < 8; ++r) u[r] = *(const unsigned int*)(rp + ((r & 3) << 1) + ((r >> 2) << 4));
  return __builtin_bit_cast(v16bf, u);
}

#define WMMA_BF16(a, b, c) \
  __builtin_amdgcn_wmma_f32_16x16x32_bf16(false, (a), false, (b), (short)0, (c), false, false)

#define TT 128            // tokens per block
#define NPIX 65536        // 32 * 2048 image pixels == token count

// ---------------------------------------------------------------------------
// K1: fc1 (768->64) + 8-stage 64x64 conv chain (+residual after stage 5) + GELU
// fc1 K-loop is software-pipelined: double-buffered LDS chunks; chunk k+1's
// global loads are staged in registers while chunk k is consumed by WMMA.
// ---------------------------------------------------------------------------
__global__ __launch_bounds__(256) void k1_fc1_conv_gelu(
    const float* __restrict__ x, const float* __restrict__ fc1_w,
    const float* __restrict__ fc1_b, const float* __restrict__ conv_w,
    const float* __restrict__ conv_b, float* __restrict__ xs_out) {
  __shared__ unsigned short s_cw[8 * 64 * 66];   // conv weights bf16 [st][o][c], 67.6KB
  __shared__ unsigned short s_x[2 * TT * 66];    // x K-chunk bf16, double buffer, 33.8KB
  __shared__ unsigned short s_w1[2 * 64 * 66];   // fc1 K-chunk bf16, double buffer, 16.9KB
  __shared__ unsigned short s_act[TT * 66];      // activations bf16, 16.9KB

  const int tid = threadIdx.x, lane = tid & 31, wave = tid >> 5;
  const int tb = blockIdx.x * TT;

  // conv weights: 8*64*32 float2 pairs -> packed u32 in LDS
  unsigned int* s_cw32 = (unsigned int*)s_cw;
  for (int i = tid; i < 8 * 64 * 32; i += 256) {
    int st = i >> 11, rem = i & 2047, o = rem >> 5, c2 = rem & 31;
    float2 f = *(const float2*)(conv_w + st * 4096 + o * 64 + (c2 << 1));
    s_cw32[st * 2112 + o * 33 + c2] = packbf2(f.x, f.y);
  }

  v8f acc[4];
#pragma unroll
  for (int n = 0; n < 4; ++n)
#pragma unroll
    for (int r = 0; r < 8; ++r) acc[n][r] = 0.0f;

  // per-thread cooperative-load coordinates (fixed across iterations)
  const int xt = tid >> 5, xc2 = tid & 31;   // + i*8 rows for x, + i*8 rows for w

  unsigned int* s_x32 = (unsigned int*)s_x;
  unsigned int* s_w132 = (unsigned int*)s_w1;

  // prologue: stage chunk 0 in registers
  float2 fx[16], fw[8];
#pragma unroll
  for (int i = 0; i < 16; ++i)
    fx[i] = *(const float2*)(x + (size_t)(tb + xt + i * 8) * 768 + (xc2 << 1));
#pragma unroll
  for (int i = 0; i < 8; ++i)
    fw[i] = *(const float2*)(fc1_w + (xt + i * 8) * 768 + (xc2 << 1));

#pragma unroll 1
  for (int it = 0; it < 12; ++it) {
    const int buf = it & 1;
    // pack + store staged chunk into LDS buffer `buf`
    unsigned int* dx = s_x32 + buf * (TT * 33);
    unsigned int* dw = s_w132 + buf * (64 * 33);
#pragma unroll
    for (int i = 0; i < 16; ++i) dx[(xt + i * 8) * 33 + xc2] = packbf2(fx[i].x, fx[i].y);
#pragma unroll
    for (int i = 0; i < 8; ++i) dw[(xt + i * 8) * 33 + xc2] = packbf2(fw[i].x, fw[i].y);
    __syncthreads();
    // stage next chunk's global loads (consumed next iteration, after compute)
    if (it < 11) {
      const int kc = (it + 1) * 64;
#pragma unroll
      for (int i = 0; i < 16; ++i)
        fx[i] = *(const float2*)(x + (size_t)(tb + xt + i * 8) * 768 + kc + (xc2 << 1));
#pragma unroll
      for (int i = 0; i < 8; ++i)
        fw[i] = *(const float2*)(fc1_w + (xt + i * 8) * 768 + kc + (xc2 << 1));
    }
    // compute from LDS buffer `buf`: load all fragments, then 8 WMMAs
    const unsigned short* bx = s_x + buf * (TT * 66) + wave * 16 * 66;
    const unsigned short* bw = s_w1 + buf * (64 * 66);
    v16bf a0 = load_a_bf16(bx, 66, lane, 0);
    v16bf a1 = load_a_bf16(bx, 66, lane, 32);
    v16bf bf[4][2];
#pragma unroll
    for (int n = 0; n < 4; ++n) {
      bf[n][0] = load_b_bf16(bw, 66, n * 16, lane, 0);
      bf[n][1] = load_b_bf16(bw, 66, n * 16, lane, 32);
    }
#pragma unroll
    for (int n = 0; n < 4; ++n) acc[n] = WMMA_BF16(a0, bf[n][0], acc[n]);
#pragma unroll
    for (int n = 0; n < 4; ++n) acc[n] = WMMA_BF16(a1, bf[n][1], acc[n]);
  }

  // bias + write bf16 activation tile (wave-private rows)
  const int row0 = wave * 16 + ((lane >> 4) << 3);
  const int nn = lane & 15;
#pragma unroll
  for (int n = 0; n < 4; ++n) {
    float bn = fc1_b[n * 16 + nn];
#pragma unroll
    for (int r = 0; r < 8; ++r)
      s_act[(row0 + r) * 66 + n * 16 + nn] = f2bf16(acc[n][r] + bn);
  }

  // ---- conv chain: 8 stages of 64x64, residual from stage 5 -> after stage 7 ----
  float res[4][8];
#pragma unroll 1
  for (int st = 0; st < 8; ++st) {
    const unsigned short* cw = s_cw + st * 4224;
    v16bf a0 = load_a_bf16(s_act + wave * 16 * 66, 66, lane, 0);
    v16bf a1 = load_a_bf16(s_act + wave * 16 * 66, 66, lane, 32);
    v16bf bf[4][2];
#pragma unroll
    for (int n = 0; n < 4; ++n) {
      bf[n][0] = load_b_bf16(cw, 66, n * 16, lane, 0);
      bf[n][1] = load_b_bf16(cw, 66, n * 16, lane, 32);
    }
    v8f d[4];
#pragma unroll
    for (int n = 0; n < 4; ++n)
#pragma unroll
      for (int r = 0; r < 8; ++r) d[n][r] = 0.0f;
#pragma unroll
    for (int n = 0; n < 4; ++n) d[n] = WMMA_BF16(a0, bf[n][0], d[n]);
#pragma unroll
    for (int n = 0; n < 4; ++n) d[n] = WMMA_BF16(a1, bf[n][1], d[n]);
#pragma unroll
    for (int n = 0; n < 4; ++n) {
      float bn = conv_b[st * 64 + n * 16 + nn];
#pragma unroll
      for (int r = 0; r < 8; ++r) {
        float v = d[n][r] + bn;
        if (st == 5) res[n][r] = v;
        if (st == 7) {
          v += res[n][r];
          float g = 0.5f * v * (1.0f + erff(v * 0.70710678118654752f));  // exact GELU
          xs_out[(size_t)(tb + row0 + r) * 64 + n * 16 + nn] = g;
        } else {
          s_act[(row0 + r) * 66 + n * 16 + nn] = f2bf16(v);
        }
      }
    }
  }
}

// ---------------------------------------------------------------------------
// K2a: per-pixel qkv matvec 64 -> 48 (image planes; perfectly coalesced)
// ---------------------------------------------------------------------------
__global__ __launch_bounds__(256) void k2a_qkv(
    const float* __restrict__ xs, const float* __restrict__ qkv_w,
    const float* __restrict__ qkv_b, float* __restrict__ qkv) {
  __shared__ float s_w[48 * 64];
  __shared__ float s_b[48];
  int tid = threadIdx.x;
  for (int i = tid; i < 48 * 64; i += 256) s_w[i] = qkv_w[i];
  if (tid < 48) s_b[tid] = qkv_b[tid];
  __syncthreads();
  int p = blockIdx.x * 256 + tid;
  float a[48];
#pragma unroll
  for (int o = 0; o < 48; ++o) a[o] = 0.0f;
#pragma unroll 1
  for (int c = 0; c < 64; ++c) {
    float xv = xs[c * NPIX + p];
#pragma unroll
    for (int o = 0; o < 48; ++o) a[o] = fmaf(s_w[o * 64 + c], xv, a[o]);
  }
#pragma unroll
  for (int o = 0; o < 48; ++o) qkv[o * NPIX + p] = a[o] + s_b[o];
}

// ---------------------------------------------------------------------------
// K2b: neighborhood attention (shift + learned 3x3 dwconv on k/v, rpb,
//      softmax over 9, weighted v) fused with proj 16 -> 64
// ---------------------------------------------------------------------------
__global__ __launch_bounds__(256) void k2b_attn_proj(
    const float* __restrict__ qkv, const float* __restrict__ dep1_w,
    const float* __restrict__ dep_b, const float* __restrict__ dep1_b,
    const float* __restrict__ rpb, const float* __restrict__ proj_w,
    const float* __restrict__ proj_b, float* __restrict__ Obuf) {
  __shared__ float s_dw[18 * 9], s_db[18], s_rpb[72], s_pw[64 * 16], s_pb[64];
  int tid = threadIdx.x;
  for (int i = tid; i < 18 * 9; i += 256) s_dw[i] = dep1_w[i];
  if (tid < 18) s_db[tid] = dep_b[tid] + dep1_b[tid];
  if (tid < 72) s_rpb[tid] = rpb[tid];
  for (int i = tid; i < 64 * 16; i += 256) s_pw[i] = proj_w[i];
  if (tid < 64) s_pb[tid] = proj_b[tid];
  __syncthreads();

  const float SCALE = 0.35355339059327373f;  // (64/8)^-0.5
  int p = blockIdx.x * 256 + tid;
  int h = p >> 11, w = p & 2047;
  float o16[16];
#pragma unroll
  for (int n = 0; n < 8; ++n) {
    const float* qp = qkv + (size_t)n * 6 * NPIX;
    float q0 = qp[p], q1 = qp[NPIX + p];
    float kk[2][9], vv[2][9];
#pragma unroll
    for (int j = 0; j < 9; ++j) {
      int dy = j / 3 - 1, dx = j % 3 - 1;
      int hh = h + dy, ww = w + dx;
      bool ok = ((unsigned)hh < 32u) && ((unsigned)ww < 2048u);
      int q = hh * 2048 + ww;
#pragma unroll
      for (int c = 0; c < 2; ++c) {
        kk[c][j] = ok ? qp[(2 + c) * NPIX + q] : 0.0f;
        vv[c][j] = ok ? qp[(4 + c) * NPIX + q] : 0.0f;
      }
    }
    float lg[9], v6[2][9];
#pragma unroll
    for (int j = 0; j < 9; ++j) {
      float l = 0.0f;
#pragma unroll
      for (int c = 0; c < 2; ++c) {
        int oc = c * 9 + j;
        float ck = s_db[oc], cv = s_db[oc];
#pragma unroll
        for (int t = 0; t < 9; ++t) {
          float wgt = s_dw[oc * 9 + t];
          ck = fmaf(wgt, kk[c][t], ck);
          cv = fmaf(wgt, vv[c][t], cv);
        }
        float k6 = kk[c][j] + ck + s_rpb[n * 9 + j];
        v6[c][j] = vv[c][j] + cv;
        float qc = (c == 0) ? q0 : q1;
        l = fmaf(qc, k6, l);
      }
      lg[j] = l * SCALE;
    }
    float m = lg[0];
#pragma unroll
    for (int j = 1; j < 9; ++j) m = fmaxf(m, lg[j]);
    float s = 0.0f, e[9];
#pragma unroll
    for (int j = 0; j < 9; ++j) { e[j] = expf(lg[j] - m); s += e[j]; }
    float inv = 1.0f / s;
#pragma unroll
    for (int c = 0; c < 2; ++c) {
      float acc2 = 0.0f;
#pragma unroll
      for (int j = 0; j < 9; ++j) acc2 = fmaf(e[j], v6[c][j], acc2);
      o16[n * 2 + c] = acc2 * inv;
    }
  }
#pragma unroll 1
  for (int cp = 0; cp < 64; ++cp) {
    float s = s_pb[cp];
#pragma unroll
    for (int i = 0; i < 16; ++i) s = fmaf(s_pw[cp * 16 + i], o16[i], s);
    Obuf[cp * NPIX + p] = s;
  }
}

// ---------------------------------------------------------------------------
// K3: fc2 (64 -> 768) + residual with x; two independent WMMA chains per iter
// ---------------------------------------------------------------------------
__global__ __launch_bounds__(256) void k3_fc2_res(
    const float* __restrict__ Oin, const float* __restrict__ fc2_w,
    const float* __restrict__ fc2_b, const float* __restrict__ x,
    float* __restrict__ out) {
  __shared__ unsigned short s_w2[768 * 66];   // fc2_w bf16, 101.4KB
  __shared__ unsigned short s_act[TT * 66];   // act bf16, 16.9KB
  int tid = threadIdx.x, lane = tid & 31, wave = tid >> 5;
  int tb = blockIdx.x * TT;
  unsigned int* s_w232 = (unsigned int*)s_w2;
  for (int i = tid; i < 768 * 32; i += 256) {   // 768 rows * 32 pairs
    int r = i >> 5, c2 = i & 31;
    float2 f = *(const float2*)(fc2_w + r * 64 + (c2 << 1));
    s_w232[r * 33 + c2] = packbf2(f.x, f.y);
  }
  unsigned int* s_act32 = (unsigned int*)s_act;
#pragma unroll
  for (int i = 0; i < 16; ++i) {                // 128 rows * 32 pairs / 256
    int idx = tid + (i << 8);
    int t = idx >> 5, c2 = idx & 31;
    float2 f = *(const float2*)(Oin + (size_t)(tb + t) * 64 + (c2 << 1));
    s_act32[t * 33 + c2] = packbf2(f.x, f.y);   // flat reinterpretation img->token
  }
  __syncthreads();
  v16bf a0 = load_a_bf16(s_act + wave * 16 * 66, 66, lane, 0);
  v16bf a1 = load_a_bf16(s_act + wave * 16 * 66, 66, lane, 32);
  const int row0 = wave * 16 + ((lane >> 4) << 3);
  const int nn = lane & 15;
#pragma unroll 1
  for (int n2 = 0; n2 < 24; ++n2) {
    const int n0 = n2 * 2, n1 = n0 + 1;
    v16bf b00 = load_b_bf16(s_w2, 66, n0 * 16, lane, 0);
    v16bf b01 = load_b_bf16(s_w2, 66, n0 * 16, lane, 32);
    v16bf b10 = load_b_bf16(s_w2, 66, n1 * 16, lane, 0);
    v16bf b11 = load_b_bf16(s_w2, 66, n1 * 16, lane, 32);
    v8f d0, d1;
#pragma unroll
    for (int r = 0; r < 8; ++r) { d0[r] = 0.0f; d1[r] = 0.0f; }
    d0 = WMMA_BF16(a0, b00, d0);
    d1 = WMMA_BF16(a0, b10, d1);
    d0 = WMMA_BF16(a1, b01, d0);
    d1 = WMMA_BF16(a1, b11, d1);
    float bn0 = fc2_b[n0 * 16 + nn];
    float bn1 = fc2_b[n1 * 16 + nn];
#pragma unroll
    for (int r = 0; r < 8; ++r) {
      size_t base = (size_t)(tb + row0 + r) * 768;
      size_t i0 = base + n0 * 16 + nn;
      size_t i1 = base + n1 * 16 + nn;
      out[i0] = x[i0] + d0[r] + bn0;
      out[i1] = x[i1] + d1[r] + bn1;
    }
  }
}

// ---------------------------------------------------------------------------
extern "C" void kernel_launch(void* const* d_in, const int* in_sizes, int n_in,
                              void* d_out, int out_size, void* d_ws, size_t ws_size,
                              hipStream_t stream) {
  const float* x      = (const float*)d_in[0];
  const float* fc1_w  = (const float*)d_in[1];
  const float* fc1_b  = (const float*)d_in[2];
  const float* conv_w = (const float*)d_in[3];
  const float* conv_b = (const float*)d_in[4];
  const float* qkv_w  = (const float*)d_in[5];
  const float* qkv_b  = (const float*)d_in[6];
  const float* dep1_w = (const float*)d_in[7];
  const float* dep_b  = (const float*)d_in[8];
  const float* dep1_b = (const float*)d_in[9];
  const float* rpb    = (const float*)d_in[10];
  const float* proj_w = (const float*)d_in[11];
  const float* proj_b = (const float*)d_in[12];
  const float* fc2_w  = (const float*)d_in[13];
  const float* fc2_b  = (const float*)d_in[14];

  float* ws   = (float*)d_ws;
  float* xs   = ws;                         // 64 * 65536 floats (16 MB)
  float* qkvb = ws + (size_t)64 * NPIX;     // 48 * 65536 floats (12 MB)
  float* Obuf = ws + (size_t)112 * NPIX;    // 64 * 65536 floats (16 MB)
  float* out  = (float*)d_out;

  k1_fc1_conv_gelu<<<512, 256, 0, stream>>>(x, fc1_w, fc1_b, conv_w, conv_b, xs);
  k2a_qkv<<<256, 256, 0, stream>>>(xs, qkv_w, qkv_b, qkvb);
  k2b_attn_proj<<<256, 256, 0, stream>>>(qkvb, dep1_w, dep_b, dep1_b, rpb, proj_w, proj_b, Obuf);
  k3_fc2_res<<<512, 256, 0, stream>>>(Obuf, fc2_w, fc2_b, x, out);
}
